// PointerGen_38122129719662
// MI455X (gfx1250) — matrix operand order repaired and checked by reading, not compiled, not run on hardware
//
#include <hip/hip_runtime.h>

#define B_   8
#define T_   128
#define SENC 400
#define H_   256
#define V_   50257

typedef __attribute__((ext_vector_type(16))) _Float16 v16h;
typedef __attribute__((ext_vector_type(8)))  float    v8f;

// ---------------------------------------------------------------------------
// Kernel 1: p_gen = sigmoid(ctx@Wc + dec_out@Wo + dec_in@Wi + bc+bo+bi)
// One wave (32 lanes) per 16-row tile of the flattened [B*T, H] inputs.
// Uses v_wmma_f32_16x16x32_f16: A = 16x32 f16 data tile, B = weight vector
// broadcast across all 16 N columns (so D[m][n] == y[m] for every n).
// ---------------------------------------------------------------------------
__global__ __launch_bounds__(32) void pgen_wmma_kernel(
    const float* __restrict__ ctx, const float* __restrict__ dec_out,
    const float* __restrict__ dec_in, const float* __restrict__ Wc,
    const float* __restrict__ Wo, const float* __restrict__ Wi,
    const float* __restrict__ bc, const float* __restrict__ bo,
    const float* __restrict__ bi, float* __restrict__ pgen)
{
    const int lane = threadIdx.x & 31;
    const int half = lane >> 4;   // K half: lanes 0-15 -> K 0..7,16..23; 16-31 -> K 8..15,24..31
    const int m    = lane & 15;   // matrix row within tile
    const int row0 = blockIdx.x * 16;

    // A-operand K mapping for 16-bit 16x32 A (CDNA5 ISA 7.12.2):
    //   VGPR j (pairs): j<4 -> K = half*8 + 2j + slot ; j>=4 -> K = 16 + half*8 + 2(j-4) + slot
    int kmap[16];
#pragma unroll
    for (int i = 0; i < 16; ++i) {
        const int j = i >> 1, slot = i & 1;
        kmap[i] = (j < 4) ? (half * 8 + j * 2 + slot)
                          : (16 + half * 8 + (j - 4) * 2 + slot);
    }

    v8f acc = {};
    const float* Xs[3] = {ctx, dec_out, dec_in};
    const float* Ws[3] = {Wc, Wo, Wi};

    for (int s = 0; s < 3; ++s) {
        const float* X = Xs[s] + (size_t)(row0 + m) * H_;
        const float* W = Ws[s];
        for (int kb = 0; kb < H_; kb += 32) {
            v16h a, bmat;
#pragma unroll
            for (int i = 0; i < 16; ++i) {
                a[i]    = (_Float16)X[kb + kmap[i]];
                // B (32x16): same K mapping as A (kmap);
                // identical value for every N column -> broadcast weight.
                bmat[i] = (_Float16)W[kb + kmap[i]];
            }
            acc = __builtin_amdgcn_wmma_f32_16x16x32_f16(
                false, a, false, bmat, (short)0, acc, false, false);
        }
    }

    // D layout (32-bit 16x16): lane L, VGPR r holds D[M = r + 8*(L>=16)][N = L%16].
    // All N columns are equal, so lanes 0 and 16 emit the 16 row results.
    if (m == 0) {
        const float bsum = bc[0] + bo[0] + bi[0];
#pragma unroll
        for (int r = 0; r < 8; ++r) {
            const float x = acc[r] + bsum;
            pgen[row0 + half * 8 + r] = 1.0f / (1.0f + __expf(-x));
        }
    }
}

// ---------------------------------------------------------------------------
// Kernel 2: fused copy-dist scatter + log_softmax + gated mix.
// One workgroup per (b,t) row; the full V=50257 f32 row lives in LDS
// (201 KB, fits the 320 KB CDNA5 WGP LDS). Global traffic = vocab read +
// out write only (~412 MB total -> ~18 us at 23.3 TB/s).
// ---------------------------------------------------------------------------
__global__ __launch_bounds__(512) void fused_copy_softmax_kernel(
    const int*   __restrict__ tokens,  // [B, SENC]
    const float* __restrict__ vocab,   // [B, T, V]
    const float* __restrict__ attn,    // [B, T, SENC]
    const float* __restrict__ pgen,    // [B*T]
    float*       __restrict__ out)     // [B, T, V]
{
    __shared__ float s_copy[V_];
    __shared__ float s_red[512];

    const int bt  = blockIdx.x;
    const int b   = bt / T_;
    const int tid = threadIdx.x;
    const int nt  = blockDim.x;

    // 1) zero the row
    for (int v = tid; v < V_; v += nt) s_copy[v] = 0.0f;
    __syncthreads();

    // 2) scatter-add the 400 attention weights (ds_add_f32 atomics)
    {
        const int*   tok = tokens + b * SENC;
        const float* at  = attn + (size_t)bt * SENC;
        for (int s = tid; s < SENC; s += nt)
            atomicAdd(&s_copy[tok[s]], at[s]);
    }
    __syncthreads();

    // 3) logsumexp over the row (zeros are real entries; exp(0)=1 included)
    float lm = 0.0f;
    for (int v = tid; v < V_; v += nt) lm = fmaxf(lm, s_copy[v]);
    s_red[tid] = lm;
    __syncthreads();
    for (int off = nt >> 1; off > 0; off >>= 1) {
        if (tid < off) s_red[tid] = fmaxf(s_red[tid], s_red[tid + off]);
        __syncthreads();
    }
    const float mrow = s_red[0];
    __syncthreads();

    float ls = 0.0f;
    for (int v = tid; v < V_; v += nt) ls += __expf(s_copy[v] - mrow);
    s_red[tid] = ls;
    __syncthreads();
    for (int off = nt >> 1; off > 0; off >>= 1) {
        if (tid < off) s_red[tid] += s_red[tid + off];
        __syncthreads();
    }
    const float lse = mrow + __logf(s_red[0]);

    // 4) final mix, streamed with b128 loads/stores (alignment-peeled)
    const float  p    = pgen[bt];
    const float  q    = 1.0f - p;
    const size_t base = (size_t)bt * V_;
    const int    peel = (int)((4 - (base & 3)) & 3);
    const int    n4   = (V_ - peel) >> 2;

    const float4* vp4 = reinterpret_cast<const float4*>(vocab + base + peel);
    float4*       op4 = reinterpret_cast<float4*>(out + base + peel);
    for (int i = tid; i < n4; i += nt) {
        const float4 vd = vp4[i];
        const int    v  = peel + (i << 2);
        float4 r;
        r.x = vd.x * p + q * (s_copy[v + 0] - lse);
        r.y = vd.y * p + q * (s_copy[v + 1] - lse);
        r.z = vd.z * p + q * (s_copy[v + 2] - lse);
        r.w = vd.w * p + q * (s_copy[v + 3] - lse);
        op4[i] = r;
    }
    for (int v = tid; v < peel; v += nt)
        out[base + v] = vocab[base + v] * p + q * (s_copy[v] - lse);
    for (int v = peel + (n4 << 2) + tid; v < V_; v += nt)
        out[base + v] = vocab[base + v] * p + q * (s_copy[v] - lse);
}

// ---------------------------------------------------------------------------
extern "C" void kernel_launch(void* const* d_in, const int* in_sizes, int n_in,
                              void* d_out, int out_size, void* d_ws, size_t ws_size,
                              hipStream_t stream)
{
    const int*   tokens  = (const int*)  d_in[0];
    const float* ctx     = (const float*)d_in[1];
    const float* dec_in  = (const float*)d_in[2];
    const float* dec_out = (const float*)d_in[3];
    const float* vocab   = (const float*)d_in[4];
    const float* attn    = (const float*)d_in[5];
    // d_in[6] = encoder_outputs: unused by the reference math
    const float* Wc = (const float*)d_in[7];
    const float* bc = (const float*)d_in[8];
    const float* Wo = (const float*)d_in[9];
    const float* bo = (const float*)d_in[10];
    const float* Wi = (const float*)d_in[11];
    const float* bi = (const float*)d_in[12];

    float* out  = (float*)d_out;
    float* pgen = (float*)d_ws;   // B*T floats = 4 KB of workspace

    pgen_wmma_kernel<<<(B_ * T_) / 16, 32, 0, stream>>>(
        ctx, dec_out, dec_in, Wc, Wo, Wi, bc, bo, bi, pgen);

    fused_copy_softmax_kernel<<<B_ * T_, 512, 0, stream>>>(
        tokens, vocab, attn, pgen, out);
}
